// MultiHeadAttention_64476049048244
// MI455X (gfx1250) — matrix operand order repaired and checked by reading, not compiled, not run on hardware
//
#include <hip/hip_runtime.h>
#include <hip/hip_bf16.h>

// ---------------- problem constants ----------------
#define D_MODEL   2048
#define N_HEADSQ  16
#define KV_HEADS  4
#define HEAD_DIM  128
#define SINKS     2
#define SEQ       2048
#define BATCH     2
#define KV_LEN    (SEQ + SINKS)           // 2050
#define SM_SCALE  0.088388347648318447f   // 1/sqrt(128)
#define LOG_THETA 9.2103403719761836f     // ln(10000)

typedef __bf16 bf16_t;
typedef __bf16 v8bf  __attribute__((ext_vector_type(8)));
typedef __bf16 v16bf __attribute__((ext_vector_type(16)));
typedef float  v8f   __attribute__((ext_vector_type(8)));
typedef int    v4i   __attribute__((ext_vector_type(4)));

#define MYMIN(a,b) ((a) < (b) ? (a) : (b))

// ---- CDNA5 async global->LDS path (signature confirmed by round-2 diagnostic:
//      (int4 AS1*, int4 AS3*, imm offset, imm cpol)) ----
#if defined(__AMDGCN__) && __has_builtin(__builtin_amdgcn_global_load_async_to_lds_b128)
#define HAVE_ASYNC_LDS 1
typedef __attribute__((address_space(1))) v4i g1_v4i;
typedef __attribute__((address_space(3))) v4i l3_v4i;
#define GASYNC_B128(gp, lp)                                                  \
    __builtin_amdgcn_global_load_async_to_lds_b128(                          \
        (g1_v4i*)(gp), (l3_v4i*)(lp), 0, 0)
#else
#define HAVE_ASYNC_LDS 0
#endif

#if HAVE_ASYNC_LDS
__device__ __forceinline__ void wait_async0() {
#if __has_builtin(__builtin_amdgcn_s_wait_asynccnt)
    __builtin_amdgcn_s_wait_asynccnt(0);
#else
    asm volatile("s_wait_asynccnt 0x0" ::: "memory");
#endif
}
#endif

// bf16 <-> f32 via bit ops (no reliance on __bf16 arithmetic)
__device__ __forceinline__ bf16_t f2bf(float f) {
    union { float f; unsigned u; } x; x.f = f;
    unsigned r = (x.u + 0x7fffu + ((x.u >> 16) & 1u)) >> 16;
    union { unsigned short s; bf16_t b; } y; y.s = (unsigned short)r;
    return y.b;
}
__device__ __forceinline__ float bf2f(bf16_t b) {
    union { unsigned short s; bf16_t b; } y; y.b = b;
    union { unsigned u; float f; } x; x.u = ((unsigned)y.s) << 16;
    return x.f;
}
__device__ __forceinline__ v16bf cat8(v8bf lo, v8bf hi) {
    return __builtin_shufflevector(lo, hi, 0,1,2,3,4,5,6,7,8,9,10,11,12,13,14,15);
}
__device__ __forceinline__ v8f zero8f() {
    v8f z = {0.f,0.f,0.f,0.f,0.f,0.f,0.f,0.f};
    return z;
}

// ---------------- fp32 -> bf16 convert ----------------
__global__ void k_cvt_bf16(const float* __restrict__ in, bf16_t* __restrict__ out, int n) {
    int i = (blockIdx.x * blockDim.x + threadIdx.x) * 4;
    if (i + 3 < n) {
        float4 f = *(const float4*)(in + i);
        out[i + 0] = f2bf(f.x);
        out[i + 1] = f2bf(f.y);
        out[i + 2] = f2bf(f.z);
        out[i + 3] = f2bf(f.w);
    }
}

// ---------------- sink K/V projection (tiny) ----------------
__global__ void k_sink_proj(const float* __restrict__ sink,
                            const float* __restrict__ Wk,
                            const float* __restrict__ Wv,
                            bf16_t* __restrict__ kbuf, bf16_t* __restrict__ vbuf) {
    int t = blockIdx.x * blockDim.x + threadIdx.x;        // 2 * SINKS * 512
    if (t >= 2 * SINKS * KV_HEADS * HEAD_DIM) return;
    int sel = t / (SINKS * KV_HEADS * HEAD_DIM);
    int rem = t % (SINKS * KV_HEADS * HEAD_DIM);
    int r = rem / (KV_HEADS * HEAD_DIM);
    int n = rem % (KV_HEADS * HEAD_DIM);
    const float* Wm = sel ? Wv : Wk;
    const float* srow = sink + (long)r * D_MODEL;
    const float* wrow = Wm + (long)n * D_MODEL;
    float acc = 0.f;
    for (int k = 0; k < D_MODEL; ++k) acc += srow[k] * wrow[k];
    int h = n >> 7, d = n & 127;
    bf16_t* dst = sel ? vbuf : kbuf;
    bf16_t v = f2bf(acc);
    for (int b = 0; b < BATCH; ++b)
        dst[(((long)(b * KV_HEADS + h)) * KV_LEN + r) * HEAD_DIM + d] = v;
}

// ---------------- RoPE (in place, bf16) ----------------
__global__ void k_rope(bf16_t* __restrict__ buf, int rows_per_head, long nrows_total, int mode) {
    long idx = (long)blockIdx.x * blockDim.x + threadIdx.x;
    if (idx >= nrows_total * 64) return;
    long rowg = idx >> 6;
    int i = (int)(idx & 63);
    int r = (int)(rowg % rows_per_head);
    int pos = (mode == 0) ? (r + SINKS) : ((r < SINKS) ? r : r - SINKS);
    bf16_t* p = buf + rowg * HEAD_DIM;
    float a = bf2f(p[i]);
    float b = bf2f(p[i + 64]);
    int j0 = i >> 1;
    float inv0 = __expf(-LOG_THETA * (float)j0 / 64.f);
    float inv1 = __expf(-LOG_THETA * (float)(j0 + 32) / 64.f);
    float c0, s0, c1, s1;
    __sincosf((float)pos * inv0, &s0, &c0);
    __sincosf((float)pos * inv1, &s1, &c1);
    p[i]      = f2bf(a * c0 - b * s0);
    p[i + 64] = f2bf(b * c1 + a * s1);
}

// ---------------- bf16 WMMA GEMM:  C[M,N] = A[M,K] @ W[N,K]^T ----------------
// 256 thr (8 waves), 128x128 tile, waves 2(m) x 4(n), wave tile 64x32.
// Double-buffered LDS; async global->LDS staging when available.
__global__ __launch_bounds__(256)
void k_gemm_bf16(const bf16_t* __restrict__ A, const bf16_t* __restrict__ W,
                 float* __restrict__ outf, bf16_t* __restrict__ outb,
                 int M, int N, int K, int seqpad, int s_off) {
    __shared__ bf16_t As[2][128 * 32];
    __shared__ bf16_t Ws[2][128 * 32];
    const int tid  = threadIdx.x;
    const int lane = tid & 31;
    const int wid  = tid >> 5;
    const int wm   = wid >> 2;        // 0..1
    const int wn   = wid & 3;         // 0..3
    const int bm   = blockIdx.y * 128;
    const int bn   = blockIdx.x * 128;
    const int lr   = lane & 15;
    const int lh   = lane >> 4;
    const int srow = tid >> 2;        // staging: 128 rows x 4 chunks of 8 elems
    const int sc   = (tid & 3) * 8;

    v8f acc[4][2];
#pragma unroll
    for (int mi = 0; mi < 4; ++mi)
#pragma unroll
        for (int ni = 0; ni < 2; ++ni) acc[mi][ni] = zero8f();

    auto issue = [&](int bi, int kk) {
#if HAVE_ASYNC_LDS
#pragma unroll
        for (int it = 0; it < 2; ++it) {
            int row = srow + it * 64;
            GASYNC_B128(&A[(long)(bm + row) * K + kk + sc], &As[bi][row * 32 + sc]);
            GASYNC_B128(&W[(long)(bn + row) * K + kk + sc], &Ws[bi][row * 32 + sc]);
        }
#else
#pragma unroll
        for (int it = 0; it < 2; ++it) {
            int row = srow + it * 64;
            *(v8bf*)&As[bi][row * 32 + sc] = *(const v8bf*)&A[(long)(bm + row) * K + kk + sc];
            *(v8bf*)&Ws[bi][row * 32 + sc] = *(const v8bf*)&W[(long)(bn + row) * K + kk + sc];
        }
        if (kk + 32 < K) {   // gfx1250 global_prefetch of next K-slab
            __builtin_prefetch(&A[(long)(bm + srow) * K + kk + 32], 0, 1);
            __builtin_prefetch(&W[(long)(bn + srow) * K + kk + 32], 0, 1);
        }
#endif
    };

    issue(0, 0);
    for (int k0 = 0; k0 < K; k0 += 32) {
        const int bi = (k0 >> 5) & 1;
#if HAVE_ASYNC_LDS
        wait_async0();                 // buffer bi fully landed in LDS
#endif
        __syncthreads();               // visible to all waves; prev readers done
        if (k0 + 32 < K) issue(bi ^ 1, k0 + 32);   // overlap with compute below

        v16bf af[4], bf[2];
#pragma unroll
        for (int mi = 0; mi < 4; ++mi) {
            const bf16_t* p = &As[bi][(wm * 64 + mi * 16 + lr) * 32 + lh * 8];
            af[mi] = cat8(*(const v8bf*)p, *(const v8bf*)(p + 16));
        }
#pragma unroll
        for (int ni = 0; ni < 2; ++ni) {
            const bf16_t* p = &Ws[bi][(wn * 32 + ni * 16 + lr) * 32 + lh * 8];
            bf[ni] = cat8(*(const v8bf*)p, *(const v8bf*)(p + 16));
        }
#pragma unroll
        for (int mi = 0; mi < 4; ++mi)
#pragma unroll
            for (int ni = 0; ni < 2; ++ni)
                acc[mi][ni] = __builtin_amdgcn_wmma_f32_16x16x32_bf16(
                    false, af[mi], false, bf[ni], (short)0, acc[mi][ni], false, false);
    }

#pragma unroll
    for (int mi = 0; mi < 4; ++mi)
#pragma unroll
        for (int ni = 0; ni < 2; ++ni)
#pragma unroll
            for (int r = 0; r < 8; ++r) {
                int m = bm + wm * 64 + mi * 16 + lh * 8 + r;
                int n = bn + wn * 32 + ni * 16 + lr;
                float v = acc[mi][ni][r];
                if (outf) {
                    outf[(long)m * N + n] = v;
                } else {
                    int b = m / SEQ, s = m % SEQ;
                    int h = n >> 7, d = n & 127;
                    outb[(((long)(b * (N >> 7) + h)) * seqpad + s + s_off) * HEAD_DIM + d] = f2bf(v);
                }
            }
}

// ---------------- flash attention with sinks (bf16 WMMA, f32 accum) ----------------
// grid: (SEQ/128, N_HEADSQ, BATCH), block 256 (8 waves, each wave = 16 query rows)
__global__ __launch_bounds__(256)
void k_attn(const bf16_t* __restrict__ q, const bf16_t* __restrict__ kbuf,
            const bf16_t* __restrict__ vbuf, bf16_t* __restrict__ ctx) {
    __shared__ bf16_t Ks[32 * 128];       // K tile, row-major   (8 KB)
    __shared__ bf16_t VT[128 * 32];       // V tile, transposed  (8 KB)
    __shared__ float  Ss[8][16][32];      // per-wave scores     (16 KB)
    __shared__ bf16_t Ps[8][16][32];      // per-wave probs      (8 KB)
    __shared__ float  Alp[8][16];
    __shared__ float  Linv[8][16];

    const int tid  = threadIdx.x;
    const int lane = tid & 31;
    const int w    = tid >> 5;
    const int lr   = lane & 15;
    const int lh   = lane >> 4;
    const int qb   = blockIdx.x * 128;
    const int h    = blockIdx.y;
    const int b    = blockIdx.z;
    const int hkv  = h >> 2;              // repeat_interleave groups of 4
    const long qbase = ((long)b * N_HEADSQ + h) * SEQ;
    const long kbase = ((long)b * KV_HEADS + hkv) * KV_LEN;

    // Q fragments for this wave's 16 rows (registers, reused every kv tile)
    const int sq = qb + w * 16 + lr;
    v16bf qf[4];
    {
        const bf16_t* qp = q + (qbase + sq) * HEAD_DIM;
#pragma unroll
        for (int kd = 0; kd < 4; ++kd) {
            const bf16_t* p = qp + kd * 32 + lh * 8;
            qf[kd] = cat8(*(const v8bf*)p, *(const v8bf*)(p + 16));
        }
    }

    v8f acc[8];
#pragma unroll
    for (int f = 0; f < 8; ++f) acc[f] = zero8f();
    float mrow = -1e30f, lrow = 0.f;

    const int jcount = MYMIN(KV_LEN, qb + 128 + SINKS);
    const int ntiles = (jcount + 31) >> 5;

    for (int kt = 0; kt < ntiles; ++kt) {
        __syncthreads();
        // ---- stage K tile (32 x 128, row-major). OOB rows are clamped to the
        // last valid row: those columns are masked in softmax (P == 0 exactly),
        // so duplicated-but-finite data is harmless. ----
#pragma unroll
        for (int it = 0; it < 2; ++it) {
            int vi  = tid + it * 256;        // 32 rows x 16 chunks of 8
            int row = vi >> 4, c = vi & 15;
            int jg  = kt * 32 + row;
            int jc  = MYMIN(jg, KV_LEN - 1);
#if HAVE_ASYNC_LDS
            GASYNC_B128(&kbuf[(kbase + jc) * HEAD_DIM + c * 8], &Ks[row * 128 + c * 8]);
#else
            *(v8bf*)&Ks[row * 128 + c * 8] =
                *(const v8bf*)&kbuf[(kbase + jc) * HEAD_DIM + c * 8];
#endif
        }
        // ---- stage V tile transposed (128 x 32), clamped rows ----
        {
            int kk = tid >> 3, cc = (tid & 7) * 16;
            int jg = kt * 32 + kk;
            int jc = MYMIN(jg, KV_LEN - 1);
            const bf16_t* vp = &vbuf[(kbase + jc) * HEAD_DIM + cc];
            bf16_t tmp[16];
#pragma unroll
            for (int e = 0; e < 16; ++e) tmp[e] = vp[e];
#pragma unroll
            for (int e = 0; e < 16; ++e) VT[(cc + e) * 32 + kk] = tmp[e];
        }
#if HAVE_ASYNC_LDS
        wait_async0();
#endif
        __syncthreads();

        // ---- scores: 16 x 32, two 16-col WMMA subtiles over d (K=128) ----
#pragma unroll
        for (int nt = 0; nt < 2; ++nt) {
            v8f scf = zero8f();
#pragma unroll
            for (int kd = 0; kd < 4; ++kd) {
                const bf16_t* p = &Ks[(nt * 16 + lr) * 128 + kd * 32 + lh * 8];
                v16bf kf = cat8(*(const v8bf*)p, *(const v8bf*)(p + 16));
                scf = __builtin_amdgcn_wmma_f32_16x16x32_bf16(
                    false, qf[kd], false, kf, (short)0, scf, false, false);
            }
#pragma unroll
            for (int r = 0; r < 8; ++r)
                Ss[w][lh * 8 + r][nt * 16 + lr] = scf[r];
        }

        // ---- online softmax: lanes 0..15 each own one query row ----
        if (lane < 16) {
            int srow = lane;
            int sqr  = qb + w * 16 + srow;
            float mx = mrow;
            float sv[32];
#pragma unroll
            for (int c = 0; c < 32; ++c) {
                int jg = kt * 32 + c;
                float s = Ss[w][srow][c] * SM_SCALE;
                bool vis = (jg < KV_LEN) && ((jg < SINKS) || (jg - SINKS <= sqr));
                s = vis ? s : -1e30f;
                sv[c] = s;
                mx = fmaxf(mx, s);
            }
            float alpha = __expf(mrow - mx);
            float tsum = 0.f;
#pragma unroll
            for (int c = 0; c < 32; ++c) {
                float p = __expf(sv[c] - mx);
                tsum += p;
                Ps[w][srow][c] = f2bf(p);
            }
            lrow = lrow * alpha + tsum;
            mrow = mx;
            Alp[w][srow] = alpha;
        }
        // ---- rescale accumulators (same-wave LDS, in-order) ----
#pragma unroll
        for (int f = 0; f < 8; ++f)
#pragma unroll
            for (int r = 0; r < 8; ++r)
                acc[f][r] *= Alp[w][lh * 8 + r];

        // ---- P @ V : one A fragment (16x32), 8 dv subtiles ----
        {
            const bf16_t* pp = &Ps[w][lr][lh * 8];
            v16bf pf = cat8(*(const v8bf*)pp, *(const v8bf*)(pp + 16));
#pragma unroll
            for (int f = 0; f < 8; ++f) {
                const bf16_t* vp = &VT[(f * 16 + lr) * 32 + lh * 8];
                v16bf vf = cat8(*(const v8bf*)vp, *(const v8bf*)(vp + 16));
                acc[f] = __builtin_amdgcn_wmma_f32_16x16x32_bf16(
                    false, pf, false, vf, (short)0, acc[f], false, false);
            }
        }
    }

    // ---- finalize: divide by l, store ctx in [b][s][h*128+d] (ready for O-GEMM) ----
    if (lane < 16) Linv[w][lane] = 1.f / lrow;
#pragma unroll
    for (int f = 0; f < 8; ++f)
#pragma unroll
        for (int r = 0; r < 8; ++r) {
            int s = qb + w * 16 + lh * 8 + r;
            float v = acc[f][r] * Linv[w][lh * 8 + r];
            ctx[((long)b * SEQ + s) * D_MODEL + h * HEAD_DIM + f * 16 + lr] = f2bf(v);
        }
}

// ---------------- host launcher ----------------
extern "C" void kernel_launch(void* const* d_in, const int* in_sizes, int n_in,
                              void* d_out, int out_size, void* d_ws, size_t ws_size,
                              hipStream_t stream) {
    const float* hidden = (const float*)d_in[0];
    // d_in[1] = attention_mask (causal; reconstructed analytically in-kernel)
    const float* Wq   = (const float*)d_in[2];
    const float* Wk   = (const float*)d_in[3];
    const float* Wv   = (const float*)d_in[4];
    const float* Wo   = (const float*)d_in[5];
    const float* sink = (const float*)d_in[6];
    float* out = (float*)d_out;

    char* ws = (char*)d_ws;
    size_t off = 0;
    auto take = [&](size_t bytes) -> char* {
        char* p = ws + off;
        off += (bytes + 255) & ~(size_t)255;
        return p;
    };
    const size_t M = (size_t)BATCH * SEQ;                       // 4096
    bf16_t* hsb  = (bf16_t*)take(M * D_MODEL * 2);
    bf16_t* wqb  = (bf16_t*)take((size_t)D_MODEL * D_MODEL * 2);
    bf16_t* wkb  = (bf16_t*)take((size_t)512 * D_MODEL * 2);
    bf16_t* wvb  = (bf16_t*)take((size_t)512 * D_MODEL * 2);
    bf16_t* wob  = (bf16_t*)take((size_t)D_MODEL * D_MODEL * 2);
    bf16_t* qbuf = (bf16_t*)take((size_t)BATCH * N_HEADSQ * SEQ * HEAD_DIM * 2);
    bf16_t* kbuf = (bf16_t*)take((size_t)BATCH * KV_HEADS * KV_LEN * HEAD_DIM * 2);
    bf16_t* vbuf = (bf16_t*)take((size_t)BATCH * KV_HEADS * KV_LEN * HEAD_DIM * 2);
    bf16_t* ctxb = (bf16_t*)take(M * D_MODEL * 2);

    auto cvt = [&](const float* src, bf16_t* dst, long n) {
        int blocks = (int)(n / 4 / 256);
        k_cvt_bf16<<<blocks, 256, 0, stream>>>(src, dst, (int)n);
    };
    cvt(hidden, hsb, (long)M * D_MODEL);
    cvt(Wq, wqb, (long)D_MODEL * D_MODEL);
    cvt(Wk, wkb, (long)512 * D_MODEL);
    cvt(Wv, wvb, (long)512 * D_MODEL);
    cvt(Wo, wob, (long)D_MODEL * D_MODEL);

    k_sink_proj<<<8, 256, 0, stream>>>(sink, Wk, Wv, kbuf, vbuf);

    // Q/K/V projections (bf16 scatter into head layouts)
    k_gemm_bf16<<<dim3(D_MODEL / 128, (int)(M / 128)), 256, 0, stream>>>(
        hsb, wqb, nullptr, qbuf, (int)M, D_MODEL, D_MODEL, SEQ, 0);
    k_gemm_bf16<<<dim3(512 / 128, (int)(M / 128)), 256, 0, stream>>>(
        hsb, wkb, nullptr, kbuf, (int)M, 512, D_MODEL, KV_LEN, SINKS);
    k_gemm_bf16<<<dim3(512 / 128, (int)(M / 128)), 256, 0, stream>>>(
        hsb, wvb, nullptr, vbuf, (int)M, 512, D_MODEL, KV_LEN, SINKS);

    // RoPE (in place)
    {
        long nrq = (long)BATCH * N_HEADSQ * SEQ;
        long tq  = nrq * 64;
        k_rope<<<(int)((tq + 255) / 256), 256, 0, stream>>>(qbuf, SEQ, nrq, 0);
        long nrk = (long)BATCH * KV_HEADS * KV_LEN;
        long tk  = nrk * 64;
        k_rope<<<(int)((tk + 255) / 256), 256, 0, stream>>>(kbuf, KV_LEN, nrk, 1);
    }

    // streaming attention with sinks
    k_attn<<<dim3(SEQ / 128, N_HEADSQ, BATCH), 256, 0, stream>>>(qbuf, kbuf, vbuf, ctxb);

    // output projection -> f32
    k_gemm_bf16<<<dim3(D_MODEL / 128, (int)(M / 128)), 256, 0, stream>>>(
        ctxb, wob, out, nullptr, (int)M, D_MODEL, D_MODEL, 0, 0);
}
